// MiniLSTMParallelCell_43379169689743
// MI455X (gfx1250) — matrix operand order repaired and checked by reading, not compiled, not run on hardware
//
#include <hip/hip_runtime.h>
#include <hip/hip_bf16.h>
#include <math.h>

// ---------------------------------------------------------------------------
// minLSTM parallel cell for MI455X (gfx1250, wave32, WMMA)
//   B=8, T=4096, Din=Dout=1024.  Output [B, T+1, D] fp32.
// Pipeline:
//   K1a: split Wi/Wf/Wh fp32 -> (bf16 hi, bf16 lo)            (12.6 MB ws)
//   K1b: split x fp32 -> (bf16 hi, bf16 lo) planes            (128 MB ws)
//   K2 : fused 3-gate GEMM via split-bf16 WMMA (hi*hi+hi*lo+lo*hi, fp32 acc),
//        2 M-tiles per wave sharing B fragments, + softplus/log_g epilogue
//        -> log_f, log_values                                  (2x128 MB ws)
//   K3 : per-(b,t) cumsum over D (block scan) -> a_star (staged in d_out),
//        w = log_values - a_star (in place)
//   K4a/4b/4c: chunked logcumsumexp over time (assoc. logaddexp scan),
//        out = exp(a_star + scan)
// ---------------------------------------------------------------------------

typedef __bf16 bf16_t;
typedef __attribute__((ext_vector_type(16))) __bf16 v16bf;
typedef __attribute__((ext_vector_type(8)))  __bf16 v8bf;
typedef __attribute__((ext_vector_type(8)))  float  v8f;
typedef __attribute__((ext_vector_type(4)))  float  v4f;

#define BB   8
#define TT   4096
#define DD   1024
#define MM   (BB * TT)          // 32768 rows
#define NCH  65                 // ceil(4097 / 64) time chunks
#define CHSZ 64

__device__ __forceinline__ float softplus_f(float x) {
    return fmaxf(x, 0.0f) + log1pf(expf(-fabsf(x)));
}
__device__ __forceinline__ float log_g_f(float x) {
    return (x >= 0.0f) ? logf(x + 0.5f) : -softplus_f(-x);
}
__device__ __forceinline__ float logaddexp_f(float a, float b) {
    float m = fmaxf(a, b);
    if (m == -__builtin_inff()) return -__builtin_inff();
    return m + log1pf(expf(-fabsf(a - b)));
}

// ---------------------------------------------------------------------------
// K1a: split weights fp32 -> bf16 hi/lo. wpack = [Wi_hi|Wi_lo|Wf_hi|Wf_lo|Wh_hi|Wh_lo]
// ---------------------------------------------------------------------------
__global__ void k1a_split_weights(const float* __restrict__ Wi,
                                  const float* __restrict__ Wf,
                                  const float* __restrict__ Wh,
                                  bf16_t* __restrict__ wpack) {
    int i = blockIdx.x * blockDim.x + threadIdx.x;
    const int N = DD * DD;
    if (i >= N) return;
    float v = Wi[i];
    bf16_t h = (bf16_t)v;
    wpack[0 * (size_t)N + i] = h;
    wpack[1 * (size_t)N + i] = (bf16_t)(v - (float)h);
    v = Wf[i];
    h = (bf16_t)v;
    wpack[2 * (size_t)N + i] = h;
    wpack[3 * (size_t)N + i] = (bf16_t)(v - (float)h);
    v = Wh[i];
    h = (bf16_t)v;
    wpack[4 * (size_t)N + i] = h;
    wpack[5 * (size_t)N + i] = (bf16_t)(v - (float)h);
}

// ---------------------------------------------------------------------------
// K1b: split activations x fp32 -> bf16 hi/lo planes (each [MM, DD]).
// One thread converts 4 elements (b128 in, 2x b64 out).
// ---------------------------------------------------------------------------
__global__ void k1b_split_x(const float* __restrict__ x,
                            bf16_t* __restrict__ xhi,
                            bf16_t* __restrict__ xlo) {
    size_t i4 = (size_t)(blockIdx.x) * blockDim.x + threadIdx.x;
    if (i4 >= (size_t)MM * DD / 4) return;
    size_t i = i4 * 4;
    v4f v = *(const v4f*)(x + i);
    bf16_t h0 = (bf16_t)v.x, h1 = (bf16_t)v.y, h2 = (bf16_t)v.z, h3 = (bf16_t)v.w;
    bf16_t hv[4] = { h0, h1, h2, h3 };
    bf16_t lv[4] = { (bf16_t)(v.x - (float)h0), (bf16_t)(v.y - (float)h1),
                     (bf16_t)(v.z - (float)h2), (bf16_t)(v.w - (float)h3) };
    *(uint64_t*)(xhi + i) = *(const uint64_t*)hv;
    *(uint64_t*)(xlo + i) = *(const uint64_t*)lv;
}

// ---------------------------------------------------------------------------
// K2: fused 3-gate GEMM (split-bf16 WMMA) + pointwise epilogue.
// Each wave: TWO 16(M)x16(N) tiles sharing the 6 weight fragments.
// grid = (N/16, Mtiles/16), block = 256 (8 waves -> 16 M-tiles per block).
// ---------------------------------------------------------------------------
__device__ __forceinline__ v16bf cat8(v8bf lo, v8bf hi) {
    return __builtin_shufflevector(lo, hi, 0, 1, 2, 3, 4, 5, 6, 7,
                                           8, 9, 10, 11, 12, 13, 14, 15);
}

__global__ void __launch_bounds__(256)
k2_gemm_pointwise(const bf16_t* __restrict__ xhi,
                  const bf16_t* __restrict__ xlo,
                  const bf16_t* __restrict__ wpack,
                  const float* __restrict__ Bi,
                  const float* __restrict__ Bf,
                  const float* __restrict__ Bh,
                  float* __restrict__ logf_out,
                  float* __restrict__ lv_out) {
    const int lane   = threadIdx.x & 31;
    const int wave   = threadIdx.x >> 5;
    const int mtile0 = (blockIdx.y * 8 + wave) * 2;   // two M-tiles per wave
    const int m0     = mtile0 * 16;                   // rows m0..m0+31
    const int n0     = blockIdx.x * 16;

    const int l15   = lane & 15;
    const int hi16  = lane >> 4;       // 0 | 1
    const int col   = n0 + l15;        // B-matrix col for this lane
    const int a_k8  = hi16 * 8;        // unused; A's upper half-wave also shifts K by +16 (ka)
    const int b_k16 = hi16 * 16;       // B: upper half-wave shifts K by +16

    const size_t N2 = (size_t)DD * DD;
    const bf16_t* Wi_hi = wpack + 0 * N2;
    const bf16_t* Wi_lo = wpack + 1 * N2;
    const bf16_t* Wf_hi = wpack + 2 * N2;
    const bf16_t* Wf_lo = wpack + 3 * N2;
    const bf16_t* Wh_hi = wpack + 4 * N2;
    const bf16_t* Wh_lo = wpack + 5 * N2;

    // Two A-row streams (one per M-tile)
    const bf16_t* xh0 = xhi + (size_t)(m0 + l15) * DD;
    const bf16_t* xl0 = xlo + (size_t)(m0 + l15) * DD;
    const bf16_t* xh1 = xhi + (size_t)(m0 + 16 + l15) * DD;
    const bf16_t* xl1 = xlo + (size_t)(m0 + 16 + l15) * DD;

    v8f acc_i0 = {}, acc_f0 = {}, acc_h0 = {};
    v8f acc_i1 = {}, acc_f1 = {}, acc_h1 = {};

    for (int k0 = 0; k0 < DD; k0 += 32) {
        const int ka = k0 + b_k16;
        // ---- A fragments: elems 0..15 = K ka..ka+15 (contiguous, matches B) ----
        v16bf a0h = cat8(*(const v8bf*)(xh0 + ka), *(const v8bf*)(xh0 + ka + 8));
        v16bf a0l = cat8(*(const v8bf*)(xl0 + ka), *(const v8bf*)(xl0 + ka + 8));
        v16bf a1h = cat8(*(const v8bf*)(xh1 + ka), *(const v8bf*)(xh1 + ka + 8));
        v16bf a1l = cat8(*(const v8bf*)(xl1 + ka), *(const v8bf*)(xl1 + ka + 8));
        if (k0 + 32 < DD) {
            __builtin_prefetch(xh0 + ka + 32, 0, 0);   // global_prefetch_b8
            __builtin_prefetch(xh1 + ka + 32, 0, 0);
        }

        // ---- B fragments: 16 contiguous K bf16 per lane (32B, aligned) ----
        const size_t boff = (size_t)col * DD + (size_t)(k0 + b_k16);
        v16bf bih = *(const v16bf*)(Wi_hi + boff);
        v16bf bil = *(const v16bf*)(Wi_lo + boff);
        v16bf bfh = *(const v16bf*)(Wf_hi + boff);
        v16bf bfl = *(const v16bf*)(Wf_lo + boff);
        v16bf bhh = *(const v16bf*)(Wh_hi + boff);
        v16bf bhl = *(const v16bf*)(Wh_lo + boff);

        // ---- split-bf16 products: hi*hi + hi*lo + lo*hi, both M-tiles ----
        acc_i0 = __builtin_amdgcn_wmma_f32_16x16x32_bf16(false, a0h, false, bih, (short)0, acc_i0, false, false);
        acc_i1 = __builtin_amdgcn_wmma_f32_16x16x32_bf16(false, a1h, false, bih, (short)0, acc_i1, false, false);
        acc_i0 = __builtin_amdgcn_wmma_f32_16x16x32_bf16(false, a0h, false, bil, (short)0, acc_i0, false, false);
        acc_i1 = __builtin_amdgcn_wmma_f32_16x16x32_bf16(false, a1h, false, bil, (short)0, acc_i1, false, false);
        acc_i0 = __builtin_amdgcn_wmma_f32_16x16x32_bf16(false, a0l, false, bih, (short)0, acc_i0, false, false);
        acc_i1 = __builtin_amdgcn_wmma_f32_16x16x32_bf16(false, a1l, false, bih, (short)0, acc_i1, false, false);

        acc_f0 = __builtin_amdgcn_wmma_f32_16x16x32_bf16(false, a0h, false, bfh, (short)0, acc_f0, false, false);
        acc_f1 = __builtin_amdgcn_wmma_f32_16x16x32_bf16(false, a1h, false, bfh, (short)0, acc_f1, false, false);
        acc_f0 = __builtin_amdgcn_wmma_f32_16x16x32_bf16(false, a0h, false, bfl, (short)0, acc_f0, false, false);
        acc_f1 = __builtin_amdgcn_wmma_f32_16x16x32_bf16(false, a1h, false, bfl, (short)0, acc_f1, false, false);
        acc_f0 = __builtin_amdgcn_wmma_f32_16x16x32_bf16(false, a0l, false, bfh, (short)0, acc_f0, false, false);
        acc_f1 = __builtin_amdgcn_wmma_f32_16x16x32_bf16(false, a1l, false, bfh, (short)0, acc_f1, false, false);

        acc_h0 = __builtin_amdgcn_wmma_f32_16x16x32_bf16(false, a0h, false, bhh, (short)0, acc_h0, false, false);
        acc_h1 = __builtin_amdgcn_wmma_f32_16x16x32_bf16(false, a1h, false, bhh, (short)0, acc_h1, false, false);
        acc_h0 = __builtin_amdgcn_wmma_f32_16x16x32_bf16(false, a0h, false, bhl, (short)0, acc_h0, false, false);
        acc_h1 = __builtin_amdgcn_wmma_f32_16x16x32_bf16(false, a1h, false, bhl, (short)0, acc_h1, false, false);
        acc_h0 = __builtin_amdgcn_wmma_f32_16x16x32_bf16(false, a0l, false, bhh, (short)0, acc_h0, false, false);
        acc_h1 = __builtin_amdgcn_wmma_f32_16x16x32_bf16(false, a1l, false, bhh, (short)0, acc_h1, false, false);
    }

    // ---- epilogue: bias + gate math, write log_f and log_values ----
    const float bias_i = Bi[col];
    const float bias_f = Bf[col];
    const float bias_h = Bh[col];

#pragma unroll
    for (int tile = 0; tile < 2; ++tile) {
        const v8f& ai = tile ? acc_i1 : acc_i0;
        const v8f& af = tile ? acc_f1 : acc_f0;
        const v8f& ah = tile ? acc_h1 : acc_h0;
        const int mbase = m0 + tile * 16 + hi16 * 8;  // C/D: lanes>=16 hold M=8..15
#pragma unroll
        for (int r = 0; r < 8; ++r) {
            int m = mbase + r;
            float zi = ai[r] + bias_i;
            float zf = af[r] + bias_f;
            float zh = ah[r] + bias_h;
            float diff  = softplus_f(-zf) - softplus_f(-zi);
            float log_f = -softplus_f(diff);
            float log_i = -softplus_f(-diff);
            float lvv   = log_i + log_g_f(zh);
            size_t idx = (size_t)m * DD + col;
            logf_out[idx] = log_f;
            lv_out[idx]   = lvv;
        }
    }
}

// ---------------------------------------------------------------------------
// K3: per-row (b,t) inclusive cumsum of log_f over D (1024) -> a_star.
// a_star staged into d_out[b, t+1, :]; w = lv - a_star overwrites lv.
// ---------------------------------------------------------------------------
__global__ void __launch_bounds__(256)
k3_cumsum_rows(const float* __restrict__ logf_in,
               float* __restrict__ lv_w,          // in: log_values, out: w
               float* __restrict__ out) {         // d_out, staging a_star
    __shared__ float s[256];
    const int rowid = blockIdx.x;                 // 0..B*T-1
    const int b = rowid >> 12;
    const int t = rowid & 4095;
    const size_t base = (size_t)rowid * DD;
    const int tid = threadIdx.x;

    float v[4], c[4];
    *(v4f*)&v[0] = *(const v4f*)(logf_in + base + tid * 4);
    c[0] = v[0];
#pragma unroll
    for (int j = 1; j < 4; ++j) c[j] = c[j - 1] + v[j];

    s[tid] = c[3];
    __syncthreads();
#pragma unroll
    for (int off = 1; off < 256; off <<= 1) {
        float add = (tid >= off) ? s[tid - off] : 0.0f;
        __syncthreads();
        s[tid] += add;
        __syncthreads();
    }
    float excl = (tid == 0) ? 0.0f : s[tid - 1];

    const size_t obase = ((size_t)b * (TT + 1) + (size_t)(t + 1)) * DD;
#pragma unroll
    for (int j = 0; j < 4; ++j) {
        int o = tid * 4 + j;
        float a_star = excl + c[j];
        out[obase + o] = a_star;
        lv_w[base + o] = lv_w[base + o] - a_star;
    }
}

// ---------------------------------------------------------------------------
// Time-dim logcumsumexp over L = T+1 = 4097, chunked (65 chunks of 64).
// v[0] = log_g(h0),  v[tau] = w[b, tau-1, o]  (tau>=1)
// ---------------------------------------------------------------------------
__device__ __forceinline__ float seq_val(const float* __restrict__ w,
                                         const float* __restrict__ h0,
                                         int b, int o, int tau) {
    if (tau == 0) return log_g_f(h0[(size_t)b * DD + o]);
    return w[((size_t)b * TT + (size_t)(tau - 1)) * DD + o];
}

__global__ void k4a_chunk_partials(const float* __restrict__ w,
                                   const float* __restrict__ h0,
                                   float* __restrict__ chunks) {
    int idx = blockIdx.x * blockDim.x + threadIdx.x;   // ((b*NCH)+c)*DD + o
    if (idx >= BB * NCH * DD) return;
    int o  = idx & (DD - 1);
    int bc = idx >> 10;
    int c  = bc % NCH;
    int b  = bc / NCH;
    int start = c * CHSZ;
    int end   = min(TT + 1, start + CHSZ);
    float r = -__builtin_inff();
    for (int tau = start; tau < end; ++tau)
        r = logaddexp_f(r, seq_val(w, h0, b, o, tau));
    chunks[((size_t)b * DD + o) * NCH + c] = r;
}

__global__ void k4b_chunk_scan(float* __restrict__ chunks) {
    int idx = blockIdx.x * blockDim.x + threadIdx.x;   // b*DD + o
    if (idx >= BB * DD) return;
    size_t base = (size_t)idx * NCH;
    float p = -__builtin_inff();
    for (int c = 0; c < NCH; ++c) {
        float t = chunks[base + c];
        chunks[base + c] = p;                          // exclusive prefix
        p = logaddexp_f(p, t);
    }
}

__global__ void k4c_finalize(const float* __restrict__ w,
                             const float* __restrict__ h0,
                             const float* __restrict__ chunks,
                             float* __restrict__ out) {
    int idx = blockIdx.x * blockDim.x + threadIdx.x;   // ((b*NCH)+c)*DD + o
    if (idx >= BB * NCH * DD) return;
    int o  = idx & (DD - 1);
    int bc = idx >> 10;
    int c  = bc % NCH;
    int b  = bc / NCH;
    int start = c * CHSZ;
    int end   = min(TT + 1, start + CHSZ);
    float r = chunks[((size_t)b * DD + o) * NCH + c];
    for (int tau = start; tau < end; ++tau) {
        r = logaddexp_f(r, seq_val(w, h0, b, o, tau));
        size_t oidx = ((size_t)b * (TT + 1) + (size_t)tau) * DD + o;
        float a_star = (tau == 0) ? 0.0f : out[oidx];  // staged by K3
        out[oidx] = expf(a_star + r);
    }
}

// ---------------------------------------------------------------------------
// launch
// ---------------------------------------------------------------------------
extern "C" void kernel_launch(void* const* d_in, const int* in_sizes, int n_in,
                              void* d_out, int out_size, void* d_ws, size_t ws_size,
                              hipStream_t stream) {
    const float* x  = (const float*)d_in[0];
    const float* h0 = (const float*)d_in[1];
    const float* Wi = (const float*)d_in[2];
    const float* Bi = (const float*)d_in[3];
    const float* Wf = (const float*)d_in[4];
    const float* Bf = (const float*)d_in[5];
    const float* Wh = (const float*)d_in[6];
    const float* Bh = (const float*)d_in[7];
    float* out = (float*)d_out;

    // workspace carving (~400 MB)
    char* p = (char*)d_ws;
    float*  logf_buf = (float*)p;  p += (size_t)MM * DD * sizeof(float);    // 128 MB
    float*  lv_buf   = (float*)p;  p += (size_t)MM * DD * sizeof(float);    // 128 MB
    bf16_t* xhi      = (bf16_t*)p; p += (size_t)MM * DD * sizeof(bf16_t);   // 64 MB
    bf16_t* xlo      = (bf16_t*)p; p += (size_t)MM * DD * sizeof(bf16_t);   // 64 MB
    bf16_t* wpack    = (bf16_t*)p; p += (size_t)6 * DD * DD * sizeof(bf16_t);
    float*  chunks   = (float*)p;

    // K1: precision splits
    {
        int n = DD * DD;
        k1a_split_weights<<<(n + 255) / 256, 256, 0, stream>>>(Wi, Wf, Wh, wpack);
        size_t n4 = (size_t)MM * DD / 4;
        k1b_split_x<<<(unsigned)((n4 + 255) / 256), 256, 0, stream>>>(x, xhi, xlo);
    }
    // K2: WMMA GEMM + pointwise
    {
        dim3 grid(DD / 16, (MM / 16) / 16);   // (64, 128)
        k2_gemm_pointwise<<<grid, 256, 0, stream>>>(xhi, xlo, wpack, Bi, Bf, Bh,
                                                    logf_buf, lv_buf);
    }
    // K3: feature-dim cumsum
    k3_cumsum_rows<<<MM, 256, 0, stream>>>(logf_buf, lv_buf, out);
    // K4: chunked time-dim logcumsumexp
    {
        int n = BB * NCH * DD;
        k4a_chunk_partials<<<(n + 255) / 256, 256, 0, stream>>>(lv_buf, h0, chunks);
        int m = BB * DD;
        k4b_chunk_scan<<<(m + 255) / 256, 256, 0, stream>>>(chunks);
        k4c_finalize<<<(n + 255) / 256, 256, 0, stream>>>(lv_buf, h0, chunks, out);
    }
}
